// EnigmaAttention_32280974197129
// MI455X (gfx1250) — matrix operand order repaired and checked by reading, not compiled, not run on hardware
//
#include <hip/hip_runtime.h>
#include <math.h>

// ---------------------------------------------------------------------------
// CDNA5 (gfx1250) fused attention pipeline.
// bf16 WMMA (f32 accum) + Tensor Data Mover staging + double-buffered LDS.
// ---------------------------------------------------------------------------

typedef __attribute__((ext_vector_type(16))) __bf16        v16bf;
typedef __attribute__((ext_vector_type(8)))  float         v8f;
typedef __attribute__((ext_vector_type(4)))  unsigned int  v4u;
typedef __attribute__((ext_vector_type(8)))  int           v8i;
typedef __attribute__((ext_vector_type(4)))  int           v4i;

#ifndef __has_builtin
#define __has_builtin(x) 0
#endif
#if __has_builtin(__builtin_amdgcn_tensor_load_to_lds) && \
    __has_builtin(__builtin_amdgcn_s_wait_tensorcnt)
#define USE_TDM 1
#else
#define USE_TDM 0
#endif

struct U32x8 { uint4 a, b; };

static __device__ __forceinline__ v16bf frag_from(uint4 a, uint4 b) {
    U32x8 t; t.a = a; t.b = b;
    return __builtin_bit_cast(v16bf, t);
}

static __device__ __forceinline__ unsigned short f2bf(float f) {
    unsigned u = __builtin_bit_cast(unsigned, f);
    u += 0x7FFFu + ((u >> 16) & 1u);            // round-to-nearest-even
    return (unsigned short)(u >> 16);
}
static __device__ __forceinline__ float bf2f(unsigned short h) {
    unsigned u = (unsigned)h << 16;
    return __builtin_bit_cast(float, u);
}

#if USE_TDM
// Issue one 2-D TDM tile load (bf16 elements) per calling wave.
// D# packing per CDNA5 ISA 8.3/8.4: group0 {count=1, lds_addr, global_addr,
// type=2}; group1 {data_size=1(2B), tensor_dim0/1, tile_dim0/1, dim0 stride}.
static __device__ __forceinline__ void tdm_load_2d(const unsigned short* gptr,
                                                   unsigned lds_off_bytes,
                                                   unsigned tensor_d0,
                                                   unsigned tensor_d1,
                                                   unsigned tile_d0,
                                                   unsigned tile_d1,
                                                   unsigned long long stride0) {
    const unsigned long long ga = (unsigned long long)(uintptr_t)gptr;
    v4u g0;
    g0[0] = 1u;                                          // count=1, user mode
    g0[1] = lds_off_bytes;                               // lds_addr
    g0[2] = (unsigned)(ga & 0xFFFFFFFFu);                // global_addr[31:0]
    g0[3] = (unsigned)((ga >> 32) & 0x01FFFFFFu)         // global_addr[56:32]
          | (2u << 30);                                  // type = 2 ("image")
    v8i g1;
    g1[0] = (int)(1u << 16);                             // data_size=1 -> 2 bytes
    g1[1] = (int)((tensor_d0 & 0xFFFFu) << 16);          // tensor_dim0[15:0]
    g1[2] = (int)((tensor_d0 >> 16) | ((tensor_d1 & 0xFFFFu) << 16));
    g1[3] = (int)((tensor_d1 >> 16) | (tile_d0 << 16));  // tile_dim0
    g1[4] = (int)(tile_d1 & 0xFFFFu);                    // tile_dim1, tile_dim2=0
    g1[5] = (int)(stride0 & 0xFFFFFFFFu);                // dim0_stride[31:0]
    g1[6] = (int)((stride0 >> 32) & 0xFFFFu);            // dim0_stride[47:32]
    g1[7] = 0;
    v4i z = {0, 0, 0, 0};
#if defined(__clang_major__) && __clang_major__ >= 23
    v8i z8 = {0, 0, 0, 0, 0, 0, 0, 0};
    __builtin_amdgcn_tensor_load_to_lds(g0, g1, z, z, z8, 0);
#else
    __builtin_amdgcn_tensor_load_to_lds(g0, g1, z, z, 0);
#endif
}
#endif

// ---------------- fp32 -> bf16 conversion, 4 elems / thread ----------------
__global__ __launch_bounds__(256) void cvt_kernel(const float* __restrict__ src,
                                                  unsigned short* __restrict__ dst,
                                                  int n4) {
    int i = blockIdx.x * 256 + threadIdx.x;
    if (i >= n4) return;
    const float4 v = ((const float4*)src)[i];
    ushort4 o;
    o.x = f2bf(v.x); o.y = f2bf(v.y); o.z = f2bf(v.z); o.w = f2bf(v.w);
    ((ushort4*)dst)[i] = o;
}

// ---------------- GEMM: C[m,n] = sum_k A[m,k] * B[n,k] --------------------
// A: [M,K] bf16 row-major, B: [N,K] bf16 row-major (i.e. x @ W.T).
// mode 0: fp32 [M,N] out.  mode 1: bf16 out permuted to [B,H,S,hd].
// LDS: two (A 128x32 + B 128x32) bf16 buffers, TDM-filled, double buffered.
#define BM 128
#define BN 128
#define BKK 32
#define GEMM_BUF_ELEMS ((BM + BN) * BKK)            // 8192 bf16 per buffer
#define GEMM_LDS_BYTES (2 * GEMM_BUF_ELEMS * 2)     // 32 KB

__global__ __launch_bounds__(256) void gemm_bf16(const unsigned short* __restrict__ A,
                                                 const unsigned short* __restrict__ Bw,
                                                 void* __restrict__ Cout,
                                                 int M, int N, int K,
                                                 int mode, int S, int H, int hd) {
    extern __shared__ __align__(16) unsigned short gsm[];

    const int t    = threadIdx.x;
    const int lane = t & 31, w = t >> 5;
    const int l16  = lane & 15, hlf = lane >> 4;
    const int wm   = w & 3, wn = w >> 2;        // 4(M) x 2(N) wave grid
    const int m0   = blockIdx.x * BM;
    const int n0   = blockIdx.y * BN;

    v8f acc[2][4];
    #pragma unroll
    for (int r = 0; r < 2; ++r)
        #pragma unroll
        for (int c = 0; c < 4; ++c) acc[r][c] = (v8f){0,0,0,0,0,0,0,0};

#if USE_TDM
    if (w == 0) {   // prologue: DMA first A/B tiles into buffer 0
        tdm_load_2d(A  + (size_t)m0 * K, 0u,
                    (unsigned)K, BM, BKK, BM, (unsigned long long)K);
        tdm_load_2d(Bw + (size_t)n0 * K, (unsigned)(BM * BKK * 2),
                    (unsigned)K, BN, BKK, BN, (unsigned long long)K);
    }
#else
    const int ldRow = t >> 1;
    const int ldSeg = (t & 1) * 16;
#endif

    int buf = 0;
    for (int k0 = 0; k0 < K; k0 += BKK) {
        unsigned short (*As)[BKK] =
            (unsigned short(*)[BKK])(gsm + buf * GEMM_BUF_ELEMS);
        unsigned short (*Bs)[BKK] =
            (unsigned short(*)[BKK])(gsm + buf * GEMM_BUF_ELEMS + BM * BKK);

#if USE_TDM
        if (w == 0) {
            if (k0 + BKK < K) {   // kick DMA for the next tile pair, then
                const int nxt = buf ^ 1;                 // wait only for the
                const unsigned nb = (unsigned)(nxt * GEMM_BUF_ELEMS * 2);
                tdm_load_2d(A  + (size_t)m0 * K + (k0 + BKK), nb,
                            (unsigned)K, BM, BKK, BM, (unsigned long long)K);
                tdm_load_2d(Bw + (size_t)n0 * K + (k0 + BKK), nb + BM * BKK * 2,
                            (unsigned)K, BN, BKK, BN, (unsigned long long)K);
                __builtin_amdgcn_s_wait_tensorcnt(2);    // current pair (in-order)
            } else {
                __builtin_amdgcn_s_wait_tensorcnt(0);
            }
        }
        __syncthreads();
#else
        const uint4* ga = (const uint4*)(A  + (size_t)(m0 + ldRow) * K + k0 + ldSeg);
        const uint4* gb = (const uint4*)(Bw + (size_t)(n0 + ldRow) * K + k0 + ldSeg);
        *(uint4*)&As[ldRow][ldSeg]     = ga[0];
        *(uint4*)&As[ldRow][ldSeg + 8] = ga[1];
        *(uint4*)&Bs[ldRow][ldSeg]     = gb[0];
        *(uint4*)&Bs[ldRow][ldSeg + 8] = gb[1];
        __syncthreads();
#endif

        v16bf af[2], bfr[4];
        #pragma unroll
        for (int r = 0; r < 2; ++r) {
            const int row = wm * 32 + r * 16 + l16;      // A frag: lane = M row
            af[r] = frag_from(*(const uint4*)&As[row][8 * hlf],
                              *(const uint4*)&As[row][16 + 8 * hlf]);
        }
        #pragma unroll
        for (int c = 0; c < 4; ++c) {
            const int col = wn * 64 + c * 16 + l16;      // B frag: lane = N col
            bfr[c] = frag_from(*(const uint4*)&Bs[col][8 * hlf],
                               *(const uint4*)&Bs[col][16 + 8 * hlf]);
        }
        #pragma unroll
        for (int r = 0; r < 2; ++r)
            #pragma unroll
            for (int c = 0; c < 4; ++c)
                acc[r][c] = __builtin_amdgcn_wmma_f32_16x16x32_bf16(
                    false, af[r], false, bfr[c], (short)0, acc[r][c], false, false);
        __syncthreads();
        buf ^= 1;
    }

    #pragma unroll
    for (int r = 0; r < 2; ++r)
        #pragma unroll
        for (int c = 0; c < 4; ++c)
            #pragma unroll
            for (int j = 0; j < 8; ++j) {
                const int m = m0 + wm * 32 + r * 16 + j + 8 * hlf;
                const int n = n0 + wn * 64 + c * 16 + l16;
                const float v = acc[r][c][j];
                if (mode == 0) {
                    ((float*)Cout)[(size_t)m * N + n] = v;
                } else {
                    const int b = m / S, s = m - b * S;
                    const int h = n / hd, d = n - h * hd;
                    ((unsigned short*)Cout)[(((size_t)b * H + h) * S + s) * hd + d] = f2bf(v);
                }
            }
}

// ---------------- RoPE on [B,H,S,hd] bf16 Q and K (in place) --------------
__global__ __launch_bounds__(256) void rope_kernel(unsigned short* __restrict__ q,
                                                   unsigned short* __restrict__ k,
                                                   int S, int hd, long total) {
    const long idx = (long)blockIdx.x * 256 + threadIdx.x;
    const int half = hd >> 1;
    if (idx >= total) return;
    const long row = idx / half;
    const int  j   = (int)(idx - row * half);
    const int  s   = (int)(row % S);
    const float inv = __expf(-(float)(2 * j) * (9.210340371976184f / (float)hd));
    const float ang = (float)s * inv;
    const float c = __cosf(ang), sn = __sinf(ang);
    const size_t base = (size_t)row * hd;
    {
        const float x1 = bf2f(q[base + j]), x2 = bf2f(q[base + j + half]);
        q[base + j]        = f2bf(x1 * c - x2 * sn);
        q[base + j + half] = f2bf(x2 * c + x1 * sn);
    }
    {
        const float x1 = bf2f(k[base + j]), x2 = bf2f(k[base + j + half]);
        k[base + j]        = f2bf(x1 * c - x2 * sn);
        k[base + j + half] = f2bf(x2 * c + x1 * sn);
    }
}

// ---------------- causal flash attention, bf16 WMMA, fp32 softmax ---------
#define QB  128
#define KBK 32
// dynamic LDS layout (bf16 elems): Kt[32][128] | Vt[128][32] | Pst[8*16][32]
#define ATTN_LDS_BYTES ((KBK * 128 + 128 * KBK + 8 * 16 * KBK) * 2)

__global__ __launch_bounds__(256) void attn_kernel(const unsigned short* __restrict__ qg,
                                                   const unsigned short* __restrict__ kg,
                                                   const unsigned short* __restrict__ vg,
                                                   unsigned short* __restrict__ og,
                                                   int B, int H, int S, int hd) {
    extern __shared__ __align__(16) unsigned short asmem[];
    unsigned short (*Kt)[128] = (unsigned short(*)[128])asmem;            // [key][d]
    unsigned short (*Vt)[KBK] = (unsigned short(*)[KBK])(asmem + KBK * 128); // [d][key]
    unsigned short (*Pst)[KBK] =
        (unsigned short(*)[KBK])(asmem + 2 * KBK * 128);                  // [w*16+r][key]

    const int t = threadIdx.x, lane = t & 31, w = t >> 5;
    const int l16 = lane & 15, hlf = lane >> 4;
    const int bh = blockIdx.y;
    const int b = bh / H, h = bh - b * H;
    const int q0 = blockIdx.x * QB;
    const size_t headOff = (size_t)bh * S * hd;
    const int D = H * hd;

    // Q frags for this wave's 16 rows: 4 chunks of K=32 over hd=128
    const int qrow = q0 + w * 16 + l16;
    v16bf qf[4];
    #pragma unroll
    for (int c = 0; c < 4; ++c) {
        const unsigned short* base = qg + headOff + (size_t)qrow * hd + 32 * c;
        qf[c] = frag_from(*(const uint4*)(base + 8 * hlf),
                          *(const uint4*)(base + 16 + 8 * hlf));
    }

    v8f o[8];
    #pragma unroll
    for (int i = 0; i < 8; ++i) o[i] = (v8f){0,0,0,0,0,0,0,0};
    float mrow[8], lrow[8];
    #pragma unroll
    for (int j = 0; j < 8; ++j) { mrow[j] = -3.0e38f; lrow[j] = 0.f; }

    const float scale = rsqrtf((float)hd);
    const int ldRow = t >> 3;          // 0..31 key rows
    const int ldSeg = (t & 7) * 16;    // 16-elem segment along hd
    const int kend = q0 + QB;          // causal bound for this query block

    for (int k0 = 0; k0 < kend; k0 += KBK) {
#if USE_TDM
        if (w == 0)   // DMA K block [32x128] row-major into Kt (lds offset 0)
            tdm_load_2d(kg + headOff + (size_t)k0 * hd, 0u,
                        (unsigned)hd, KBK, (unsigned)hd, KBK,
                        (unsigned long long)hd);
#else
        {
            const uint4* gk = (const uint4*)(kg + headOff + (size_t)(k0 + ldRow) * hd + ldSeg);
            *(uint4*)&Kt[ldRow][ldSeg]     = gk[0];
            *(uint4*)&Kt[ldRow][ldSeg + 8] = gk[1];
        }
#endif
        {   // V block transposed to [128x32] (manual: TDM cannot transpose)
            const unsigned short* gv = vg + headOff + (size_t)(k0 + ldRow) * hd + ldSeg;
            #pragma unroll
            for (int e = 0; e < 16; ++e) Vt[ldSeg + e][ldRow] = gv[e];
        }
#if USE_TDM
        if (w == 0) __builtin_amdgcn_s_wait_tensorcnt(0);
#endif
        __syncthreads();

        // scores S = Q*K^T : two 16x16 f32 C-frags covering 32 keys
        v8f sc[2];
        #pragma unroll
        for (int f = 0; f < 2; ++f) sc[f] = (v8f){0,0,0,0,0,0,0,0};
        #pragma unroll
        for (int f = 0; f < 2; ++f)
            #pragma unroll
            for (int c = 0; c < 4; ++c) {
                const int key = 16 * f + l16;                  // B frag: lane = key col
                v16bf kb = frag_from(*(const uint4*)&Kt[key][32 * c + 8 * hlf],
                                     *(const uint4*)&Kt[key][32 * c + 16 + 8 * hlf]);
                sc[f] = __builtin_amdgcn_wmma_f32_16x16x32_bf16(
                    false, qf[c], false, kb, (short)0, sc[f], false, false);
            }

        // scale + causal mask (C layout: row = j + 8*hlf, col = l16 + 16*f)
        #pragma unroll
        for (int f = 0; f < 2; ++f)
            #pragma unroll
            for (int j = 0; j < 8; ++j) {
                const float v = sc[f][j] * scale;
                const int qq = q0 + w * 16 + j + 8 * hlf;
                const int kk = k0 + 16 * f + l16;
                sc[f][j] = (kk > qq) ? -3.0e38f : v;
            }

        // online softmax: row reductions over 16-lane halves (wave32)
        #pragma unroll
        for (int j = 0; j < 8; ++j) {
            float rm = fmaxf(sc[0][j], sc[1][j]);
            rm = fmaxf(rm, __shfl_xor(rm, 1, 32));
            rm = fmaxf(rm, __shfl_xor(rm, 2, 32));
            rm = fmaxf(rm, __shfl_xor(rm, 4, 32));
            rm = fmaxf(rm, __shfl_xor(rm, 8, 32));
            const float mn   = fmaxf(mrow[j], rm);
            const float corr = __expf(mrow[j] - mn);
            mrow[j] = mn;
            lrow[j] *= corr;
            #pragma unroll
            for (int dch = 0; dch < 8; ++dch) o[dch][j] *= corr;
        }

        // P = exp(S-m): write C-layout to LDS, re-read in A-layout
        #pragma unroll
        for (int j = 0; j < 8; ++j) {
            float rs = 0.f;
            #pragma unroll
            for (int f = 0; f < 2; ++f) {
                const float p = __expf(sc[f][j] - mrow[j]);
                rs += p;
                Pst[w * 16 + j + 8 * hlf][16 * f + l16] = f2bf(p);
            }
            rs += __shfl_xor(rs, 1, 32);
            rs += __shfl_xor(rs, 2, 32);
            rs += __shfl_xor(rs, 4, 32);
            rs += __shfl_xor(rs, 8, 32);
            lrow[j] += rs;
        }
        // same-wave DS RAW: drain DScnt before A-layout reload
        asm volatile("s_wait_dscnt 0x0" ::: "memory");

        const v16bf pf = frag_from(*(const uint4*)&Pst[w * 16 + l16][8 * hlf],
                                   *(const uint4*)&Pst[w * 16 + l16][16 + 8 * hlf]);
        // O += P * V over hd in 8 column chunks
        #pragma unroll
        for (int dch = 0; dch < 8; ++dch) {
            const int d = 16 * dch + l16;
            v16bf vb = frag_from(*(const uint4*)&Vt[d][8 * hlf],
                                 *(const uint4*)&Vt[d][16 + 8 * hlf]);
            o[dch] = __builtin_amdgcn_wmma_f32_16x16x32_bf16(
                false, pf, false, vb, (short)0, o[dch], false, false);
        }
        __syncthreads();
    }

    // normalize and store to [B,S,H*hd] bf16 for the output projection
    #pragma unroll
    for (int j = 0; j < 8; ++j) {
        const float invl = 1.f / lrow[j];
        const int qq = q0 + w * 16 + j + 8 * hlf;
        #pragma unroll
        for (int dch = 0; dch < 8; ++dch) {
            const int d = 16 * dch + l16;
            og[((size_t)b * S + qq) * D + h * hd + d] = f2bf(o[dch][j] * invl);
        }
    }
}

// ---------------------------------------------------------------------------
extern "C" void kernel_launch(void* const* d_in, const int* in_sizes, int n_in,
                              void* d_out, int out_size, void* d_ws, size_t ws_size,
                              hipStream_t stream) {
    (void)in_sizes; (void)n_in; (void)out_size; (void)ws_size;
    const float* x  = (const float*)d_in[0];
    const float* Wq = (const float*)d_in[1];
    const float* Wk = (const float*)d_in[2];
    const float* Wv = (const float*)d_in[3];
    const float* Wo = (const float*)d_in[4];

    const int Bn = 2, S = 2048, D = 2048, H = 16, hd = 128;
    const int M = Bn * S;

    // workspace carve (bf16 elements): ~117 MB total
    unsigned short* Xb  = (unsigned short*)d_ws;
    unsigned short* Wqb = Xb  + (size_t)M * D;
    unsigned short* Wkb = Wqb + (size_t)D * D;
    unsigned short* Wvb = Wkb + (size_t)D * D;
    unsigned short* Wob = Wvb + (size_t)D * D;
    unsigned short* qw  = Wob + (size_t)D * D;
    unsigned short* kw  = qw  + (size_t)M * D;
    unsigned short* vw  = kw  + (size_t)M * D;
    unsigned short* aw  = vw  + (size_t)M * D;

    cvt_kernel<<<(M * D / 4 + 255) / 256, 256, 0, stream>>>(x, Xb, M * D / 4);
    cvt_kernel<<<(D * D / 4 + 255) / 256, 256, 0, stream>>>(Wq, Wqb, D * D / 4);
    cvt_kernel<<<(D * D / 4 + 255) / 256, 256, 0, stream>>>(Wk, Wkb, D * D / 4);
    cvt_kernel<<<(D * D / 4 + 255) / 256, 256, 0, stream>>>(Wv, Wvb, D * D / 4);
    cvt_kernel<<<(D * D / 4 + 255) / 256, 256, 0, stream>>>(Wo, Wob, D * D / 4);

    const dim3 gg(M / BM, D / BN);
    gemm_bf16<<<gg, 256, GEMM_LDS_BYTES, stream>>>(Xb, Wqb, qw, M, D, D, 1, S, H, hd);
    gemm_bf16<<<gg, 256, GEMM_LDS_BYTES, stream>>>(Xb, Wkb, kw, M, D, D, 1, S, H, hd);
    gemm_bf16<<<gg, 256, GEMM_LDS_BYTES, stream>>>(Xb, Wvb, vw, M, D, D, 1, S, H, hd);

    const long ropeTot = (long)Bn * H * S * (hd / 2);
    rope_kernel<<<(int)((ropeTot + 255) / 256), 256, 0, stream>>>(qw, kw, S, hd, ropeTot);

    attn_kernel<<<dim3(S / QB, Bn * H), 256, ATTN_LDS_BYTES, stream>>>(
        qw, kw, vw, aw, Bn, H, S, hd);

    gemm_bf16<<<gg, 256, GEMM_LDS_BYTES, stream>>>(aw, Wob, (float*)d_out, M, D, D, 0, S, H, hd);
}
